// MyModel_64587718197290
// MI455X (gfx1250) — matrix operand order repaired and checked by reading, not compiled, not run on hardware
//
#include <hip/hip_runtime.h>
#include <hip/hip_bf16.h>

// ---------------------------------------------------------------------------
// GCN layer on MI455X (gfx1250, wave32, WMMA bf16 16x16x32, f32 accumulate)
//   out[b] = fc_b + sum_{i,f} relu( s_i * ((A@Y)[i,f] + Y[i,f]) + bias[f] ) * fc_w[f]
//   where Y[j,f] = s_j * (H@W)[j,f],  s = 1/sqrt(colsum(A)+1)
// ---------------------------------------------------------------------------

#define BATCH 32
#define NN    1024
#define FF    128

typedef __attribute__((ext_vector_type(16))) __bf16          v16bf;
typedef __attribute__((ext_vector_type(8)))  float           v8f;
typedef __attribute__((ext_vector_type(8)))  unsigned short  v8u16;
typedef __attribute__((ext_vector_type(16))) unsigned short  v16u16;

static __device__ __forceinline__ unsigned short f2bf(float x) {
    // round-to-nearest-even f32 -> bf16
    unsigned u = __float_as_uint(x);
    unsigned r = ((u >> 16) & 1u) + 0x7fffu;
    return (unsigned short)((u + r) >> 16);
}
static __device__ __forceinline__ float bf2f(unsigned short b) {
    return __uint_as_float(((unsigned)b) << 16);
}

// -------- Kernel 1a: partial column sums of A (8 i-slices per batch) -------
// grid (8, BATCH) x 256 threads; thread handles 4 consecutive columns (b128)
__global__ void k_degrees_part(const float* __restrict__ A,
                               float* __restrict__ dpart) {
    const int b  = blockIdx.y;
    const int ib = blockIdx.x;                         // i-slice
    const int j4 = threadIdx.x * 4;                    // 4 columns
    const float* Ab = A + (size_t)b * NN * NN;
    float4 sum = make_float4(0.f, 0.f, 0.f, 0.f);
    const int i0 = ib * (NN / 8);
#pragma unroll 4
    for (int i = 0; i < NN / 8; ++i) {
        float4 v = *(const float4*)&Ab[(size_t)(i0 + i) * NN + j4];
        sum.x += v.x; sum.y += v.y; sum.z += v.z; sum.w += v.w;
    }
    *(float4*)&dpart[((size_t)b * 8 + ib) * NN + j4] = sum;
}

// -------- Kernel 1b: combine partials -> s = rsqrt(colsum + 1) -------------
__global__ void k_degrees_combine(const float* __restrict__ dpart,
                                  float* __restrict__ s) {
    const int b  = blockIdx.y;
    const int j4 = threadIdx.x * 4;
    float4 sum = make_float4(1.f, 1.f, 1.f, 1.f);      // + I self loop
#pragma unroll
    for (int ib = 0; ib < 8; ++ib) {                   // fixed order: deterministic
        float4 v = *(const float4*)&dpart[((size_t)b * 8 + ib) * NN + j4];
        sum.x += v.x; sum.y += v.y; sum.z += v.z; sum.w += v.w;
    }
    float4 r = make_float4(rsqrtf(sum.x), rsqrtf(sum.y),
                           rsqrtf(sum.z), rsqrtf(sum.w));
    *(float4*)&s[(size_t)b * NN + j4] = r;
}

// -------- Kernel 2: Y = s_j * (H @ W), bf16, pre-swizzled to B-fragment ----
// Yf index: ((((b*32 + kt)*8 + ft)*32 + lane)*16 + e)
//   kt = j>>5, e = j&15, lane = (f&15) + (((j>>4)&1)<<4), ft = f>>4
__global__ void k_hw_scale_swizzle(const float* __restrict__ H,
                                   const float* __restrict__ W,
                                   const float* __restrict__ s,
                                   unsigned short* __restrict__ Yf) {
    const int b  = blockIdx.y;
    const int j0 = blockIdx.x * 32;                    // 32 node-rows per block
    const int tid = threadIdx.x;                       // 256 threads
    __shared__ float Hs[32][FF];
    for (int i = tid; i < 32 * FF; i += 256) {
        int r = i >> 7, k = i & 127;
        Hs[r][k] = H[((size_t)b * NN + j0 + r) * FF + k];
    }
    __syncthreads();

    const int f  = tid & 127;                          // feature column
    const int rg = tid >> 7;                           // 0: rows 0-15, 1: rows 16-31
    float acc[16];
#pragma unroll
    for (int r = 0; r < 16; ++r) acc[r] = 0.0f;
    for (int k = 0; k < FF; ++k) {
        float wv = W[k * FF + f];
#pragma unroll
        for (int r = 0; r < 16; ++r)
            acc[r] = fmaf(Hs[rg * 16 + r][k], wv, acc[r]);
    }

    v16u16 ybuf;
#pragma unroll
    for (int r = 0; r < 16; ++r) {
        int j = j0 + rg * 16 + r;
        ybuf[r] = f2bf(s[b * NN + j] * acc[r]);        // e = r (contiguous K)
    }
    const int kt   = j0 >> 5;
    const int ft   = f >> 4;
    const int lane = (f & 15) + (rg << 4);
    *(v16u16*)&Yf[((((size_t)b * 32 + kt) * 8 + ft) * 32 + lane) * 16] = ybuf;
}

// -------- Kernel 3: Z = A @ Y (+Y), fused relu/bias/pool/head via WMMA -----
// grid (64, 32): blockIdx.x = 16-row tile, blockIdx.y = batch. 256 thr = 8 waves.
// Wave w owns feature tile ft=w (16 cols). Double-buffered 16x64 bf16 A tile:
// one barrier + two WMMAs per K=64 round.
__global__ void k_gcn_wmma(const float* __restrict__ A,
                           const unsigned short* __restrict__ Yf,
                           const float* __restrict__ s,
                           const float* __restrict__ bias,
                           const float* __restrict__ fcw,
                           float* __restrict__ partial) {
    const int b    = blockIdx.y;
    const int mt   = blockIdx.x;
    const int tid  = threadIdx.x;
    const int wave = tid >> 5;
    const int lane = tid & 31;
    const int hi   = lane >> 4;
    const int nlo  = lane & 15;

    __shared__ __align__(16) unsigned short As[2][16][64]; // 2 x 2KB bf16 tiles
    __shared__ float wsum[8];

    const float* Ab = A + (size_t)b * NN * NN + (size_t)mt * 16 * NN;

    // staging role: thread loads 4 consecutive f32 (one b128), packs 4 bf16
    const int sr = tid >> 4;                           // row 0..15
    const int sc = (tid & 15) * 4;                     // col 0,4,...,60

    auto stage = [&](int rnd, int bufi) {
        float4 a4 = *(const float4*)&Ab[(size_t)sr * NN + rnd * 64 + sc];
        unsigned lo = (unsigned)f2bf(a4.x) | ((unsigned)f2bf(a4.y) << 16);
        unsigned hh = (unsigned)f2bf(a4.z) | ((unsigned)f2bf(a4.w) << 16);
        *(uint2*)&As[bufi][sr][sc] = make_uint2(lo, hh);
    };

    v8f acc = {};
    stage(0, 0);
    __syncthreads();

    for (int rnd = 0; rnd < 16; ++rnd) {
        const int bufi = rnd & 1;
        if (rnd + 1 < 16) stage(rnd + 1, bufi ^ 1);    // prefetch next tile

#pragma unroll
        for (int half = 0; half < 2; ++half) {         // two K=32 sub-steps
            const int kb = half * 32;
            // A fragment: lane holds row m=nlo; K = e+hi*16 (+kb)
            v8u16 alo = *(const v8u16*)&As[bufi][nlo][kb + hi * 16];
            v8u16 ahi = *(const v8u16*)&As[bufi][nlo][kb + hi * 16 + 8];
            v16u16 au;
#pragma unroll
            for (int e = 0; e < 8; ++e) { au[e] = alo[e]; au[e + 8] = ahi[e]; }

            // B fragment: pre-swizzled, 32 contiguous bytes per lane
            const int kt = rnd * 2 + half;
            const unsigned short* yp =
                Yf + ((((size_t)b * 32 + kt) * 8 + wave) * 32 + lane) * 16;
            v16u16 bu = *(const v16u16*)yp;

            acc = __builtin_amdgcn_wmma_f32_16x16x32_bf16(
                false, __builtin_bit_cast(v16bf, au),
                false, __builtin_bit_cast(v16bf, bu),
                (short)0, acc, false, false);
        }
        __syncthreads();                               // single barrier per round
    }

    // Epilogue: +Y (identity term), row scale, bias, relu, pool * fc_w
    const int f  = wave * 16 + nlo;
    const float fw = fcw[f];
    const float bb = bias[f];
    float local = 0.0f;
#pragma unroll
    for (int v = 0; v < 8; ++v) {
        int row  = mt * 16 + v + hi * 8;               // C/D: m = v + hi*8
        int kt2  = row >> 5, kr = row & 31;
        int ln2  = nlo + ((kr >> 4) << 4);
        float yid = bf2f(Yf[((((size_t)b * 32 + kt2) * 8 + wave) * 32 + ln2) * 16
                            + (kr & 15)]);
        float h = fmaf(s[b * NN + row], acc[v] + yid, bb);
        h = fmaxf(h, 0.0f);
        local = fmaf(h, fw, local);
    }
    // deterministic in-block reduction (no float atomics -> replay-stable)
#pragma unroll
    for (int off = 16; off > 0; off >>= 1)
        local += __shfl_down(local, off, 32);
    if (lane == 0) wsum[wave] = local;
    __syncthreads();
    if (tid == 0) {
        float t = 0.0f;
#pragma unroll
        for (int i = 0; i < 8; ++i) t += wsum[i];
        partial[b * 64 + mt] = t;
    }
}

// -------- Kernel 4: deterministic final reduction --------------------------
__global__ void k_finalize(const float* __restrict__ partial,
                           const float* __restrict__ fcb,
                           float* __restrict__ out) {
    const int b = threadIdx.x;
    if (b < BATCH) {
        float acc = fcb[0];
        for (int mt = 0; mt < 64; ++mt) acc += partial[b * 64 + mt];
        out[b] = acc;
    }
}

extern "C" void kernel_launch(void* const* d_in, const int* in_sizes, int n_in,
                              void* d_out, int out_size, void* d_ws, size_t ws_size,
                              hipStream_t stream) {
    (void)in_sizes; (void)n_in; (void)out_size; (void)ws_size;
    const float* A    = (const float*)d_in[0];
    const float* H    = (const float*)d_in[1];
    const float* W    = (const float*)d_in[2];
    const float* bias = (const float*)d_in[3];
    const float* fcw  = (const float*)d_in[4];
    const float* fcb  = (const float*)d_in[5];
    float* out = (float*)d_out;

    char* ws = (char*)d_ws;
    float*          s       = (float*)ws;                                // 128 KB
    unsigned short* Yf      = (unsigned short*)(ws + 131072);            // 8 MB
    float*          partial = (float*)(ws + 131072 + (size_t)8*1024*1024);      // 8 KB
    float*          dpart   = (float*)(ws + 131072 + (size_t)8*1024*1024 + 8192); // 1 MB

    k_degrees_part    <<<dim3(8, BATCH),        256, 0, stream>>>(A, dpart);
    k_degrees_combine <<<dim3(1, BATCH),        256, 0, stream>>>(dpart, s);
    k_hw_scale_swizzle<<<dim3(NN / 32, BATCH),  256, 0, stream>>>(H, W, s, Yf);
    k_gcn_wmma        <<<dim3(NN / 16, BATCH),  256, 0, stream>>>(A, Yf, s, bias, fcw, partial);
    k_finalize        <<<1, 32, 0, stream>>>(partial, fcb, out);
}
